// LHBlock_32710470926847
// MI455X (gfx1250) — matrix operand order repaired and checked by reading, not compiled, not run on hardware
//
#include <hip/hip_runtime.h>
#include <hip/hip_bf16.h>

typedef __attribute__((ext_vector_type(16))) __bf16 v16bf;
typedef __attribute__((ext_vector_type(8)))  float  v8f;

#define EPS_BN 1e-5f
#define CL 128
#define CH 64
#define PXL 9216      // 96*96
#define PXH 36864     // 192*192
#define NS 144        // 12*12
#define NSP 160       // padded K for second attn GEMM

__device__ __forceinline__ __bf16 f2bf(float f) {
    unsigned u = __builtin_bit_cast(unsigned, f);
    u += 0x7FFFu + ((u >> 16) & 1u);
    unsigned short h = (unsigned short)(u >> 16);
    return __builtin_bit_cast(__bf16, h);
}

// branchless erf (Abramowitz-Stegun 7.1.26, |err| ~ 1.5e-7)
// uses raw v_rcp_f32 (approx) instead of IEEE-div sequence
__device__ __forceinline__ float erf_apx(float x) {
    float ax = fabsf(x);
    float t  = __builtin_amdgcn_rcpf(1.0f + 0.3275911f * ax);
    float y  = t * (0.254829592f + t * (-0.284496736f + t * (1.421413741f
             + t * (-1.453152027f + t * 1.061405429f))));
    float r  = 1.0f - y * __expf(-ax * ax);
    return copysignf(r, x);
}

__device__ __forceinline__ float gelu_exact(float x) {
    return 0.5f * x * (1.0f + erf_apx(x * 0.70710678118654752f));
}

__device__ __forceinline__ v8f wmma_bf16(v16bf a, v16bf b, v8f c) {
    return __builtin_amdgcn_wmma_f32_16x16x32_bf16(false, a, false, b, (short)0, c, false, false);
}

// A fragment (16x32 bf16): lanes 0-15 rows M, K=kb..kb+15;
// lanes 16-31 same rows, K=kb+16..kb+31
__device__ __forceinline__ v16bf a_frag_lds(const __bf16* W, int row0, int ldk, int kbase, int lane) {
    int m  = row0 + (lane & 15);
    int kh = (lane >> 4) << 4;
    const __bf16* wr = W + m * ldk + kbase + kh;
    v16bf a;
#pragma unroll
    for (int j = 0; j < 16; ++j) a[j] = wr[j];
    return a;
}

// B fragment (32x16 bf16): lane n holds column n (contiguous pixel),
// lanes 0-15 K=kbase..kbase+15, lanes 16-31 K=kbase+16..kbase+31
__device__ __forceinline__ v16bf b_frag_gmem(const float* Xp, long sC, int kbase, int lane) {
    int n  = lane & 15;
    int k0 = kbase + ((lane >> 4) << 4);
    v16bf b;
#pragma unroll
    for (int j = 0; j < 16; ++j) b[j] = f2bf(Xp[(long)(k0 + j) * sC + n]);
    return b;
}

__device__ __forceinline__ v16bf b_frag_gmem_bn(const float* Xp, long sC, int kbase, int lane,
                                                const float* sArr, const float* tArr) {
    int n  = lane & 15;
    int k0 = kbase + ((lane >> 4) << 4);
    v16bf b;
#pragma unroll
    for (int j = 0; j < 16; ++j) {
        int c = k0 + j;
        b[j] = f2bf(Xp[(long)c * sC + n] * sArr[c] + tArr[c]);
    }
    return b;
}

__device__ __forceinline__ v16bf b_frag_lds16(const __bf16* H, int kbase, int lane) {
    int n  = lane & 15;
    int k0 = kbase + ((lane >> 4) << 4);
    v16bf b;
#pragma unroll
    for (int j = 0; j < 16; ++j) b[j] = H[(k0 + j) * 16 + n];
    return b;
}

// ---------------- Kernel 1: x_l + MLP(x_l), Cl=128 ----------------
__global__ __launch_bounds__(256) void k_mlp_l(
    const float* __restrict__ x, const float* __restrict__ w1, const float* __restrict__ b1,
    const float* __restrict__ w2, const float* __restrict__ b2, float* __restrict__ y)
{
    __shared__ __bf16 wbuf[CL * CL];        // 32 KB, staged twice (w1 then w2)
    __shared__ __bf16 hs[8][CL * 16];       // 32 KB

    int tid = threadIdx.x;
    for (int i = tid; i < CL * CL; i += 256) wbuf[i] = f2bf(w1[i]);
    __syncthreads();

    int wave = tid >> 5, lane = tid & 31;
    int n = lane & 15, hi = lane >> 4;
    long g0 = ((long)blockIdx.x * 8 + wave) * 16;
    int b = (int)(g0 / PXL);
    int p = (int)(g0 % PXL);
    const float* xb = x + (long)b * CL * PXL + p;
    float*       yb = y + (long)b * CL * PXL + p;
    __bf16* h = hs[wave];

    // hoisted activation B fragments: each input element touched once
    v16bf bx[4];
#pragma unroll
    for (int kt = 0; kt < 4; ++kt) bx[kt] = b_frag_gmem(xb, PXL, kt * 32, lane);

    // stage 1: h = gelu(w1 * x + b1)
    for (int ot = 0; ot < 8; ++ot) {
        v8f acc;
#pragma unroll
        for (int r = 0; r < 8; ++r) acc[r] = b1[ot * 16 + r + 8 * hi];
#pragma unroll
        for (int kt = 0; kt < 4; ++kt)
            acc = wmma_bf16(a_frag_lds(wbuf, ot * 16, CL, kt * 32, lane), bx[kt], acc);
#pragma unroll
        for (int r = 0; r < 8; ++r) {
            int m = ot * 16 + r + 8 * hi;
            h[m * 16 + n] = f2bf(gelu_exact(acc[r]));
        }
    }
    __syncthreads();
    for (int i = tid; i < CL * CL; i += 256) wbuf[i] = f2bf(w2[i]);
    __syncthreads();

    v16bf bh[4];
#pragma unroll
    for (int kt = 0; kt < 4; ++kt) bh[kt] = b_frag_lds16(h, kt * 32, lane);

    // stage 2: y = x + w2 * h + b2
    for (int ot = 0; ot < 8; ++ot) {
        v8f acc;
#pragma unroll
        for (int r = 0; r < 8; ++r) acc[r] = b2[ot * 16 + r + 8 * hi];
#pragma unroll
        for (int kt = 0; kt < 4; ++kt)
            acc = wmma_bf16(a_frag_lds(wbuf, ot * 16, CL, kt * 32, lane), bh[kt], acc);
#pragma unroll
        for (int r = 0; r < 8; ++r) {
            int m = ot * 16 + r + 8 * hi;
            yb[(long)m * PXL + n] = acc[r] + xb[(long)m * PXL + n];
        }
    }
}

// ---------------- Kernel 2: relu(bn(conv1x1 128->64)) ----------------
__global__ __launch_bounds__(256) void k_conv_bn_relu(
    const float* __restrict__ x, const float* __restrict__ w,
    const float* __restrict__ g, const float* __restrict__ bb_, float* __restrict__ y)
{
    __shared__ __bf16 ws[CH * CL];          // 16 KB
    int tid = threadIdx.x;
    for (int i = tid; i < CH * CL; i += 256) ws[i] = f2bf(w[i]);
    __syncthreads();

    int wave = tid >> 5, lane = tid & 31;
    int n = lane & 15, hi = lane >> 4;
    long g0 = ((long)blockIdx.x * 8 + wave) * 16;
    int b = (int)(g0 / PXL);
    int p = (int)(g0 % PXL);
    const float* xb = x + (long)b * CL * PXL + p;
    float*       yb = y + (long)b * CH * PXL + p;
    float rs = rsqrtf(1.0f + EPS_BN);

    v16bf bx[4];
#pragma unroll
    for (int kt = 0; kt < 4; ++kt) bx[kt] = b_frag_gmem(xb, PXL, kt * 32, lane);

    for (int ot = 0; ot < 4; ++ot) {
        v8f acc = {0.f, 0.f, 0.f, 0.f, 0.f, 0.f, 0.f, 0.f};
#pragma unroll
        for (int kt = 0; kt < 4; ++kt)
            acc = wmma_bf16(a_frag_lds(ws, ot * 16, CL, kt * 32, lane), bx[kt], acc);
#pragma unroll
        for (int r = 0; r < 8; ++r) {
            int m = ot * 16 + r + 8 * hi;
            float v = acc[r] * (g[m] * rs) + bb_[m];
            yb[(long)m * PXL + n] = fmaxf(v, 0.0f);
        }
    }
}

// ---------------- Kernel 3: x_h + bilinear_up2x(x_l_conv) ----------------
__global__ __launch_bounds__(256) void k_up_add(
    const float* __restrict__ xh, const float* __restrict__ xlc, float* __restrict__ out)
{
    long i = (long)blockIdx.x * 256 + threadIdx.x;
    const long total = (long)8 * CH * PXH;
    if (i >= total) return;
    int xp = (int)(i % 192);
    long t = i / 192;
    int yp = (int)(t % 192);
    long bc = t / 192;
    float sy = 0.5f * yp - 0.25f;
    float sx = 0.5f * xp - 0.25f;
    int y0 = (int)floorf(sy); float fy = sy - (float)y0;
    int x0 = (int)floorf(sx); float fx = sx - (float)x0;
    int ya = max(y0, 0), yb = min(y0 + 1, 95);
    int xa = max(x0, 0), xb = min(x0 + 1, 95);
    const float* s = xlc + bc * PXL;
    float v = (1.f - fy) * ((1.f - fx) * s[ya * 96 + xa] + fx * s[ya * 96 + xb])
            +        fy  * ((1.f - fx) * s[yb * 96 + xa] + fx * s[yb * 96 + xb]);
    out[i] = xh[i] + v;
}

// ---------------- Kernel 4a: bn(kv) + 8x8 max-pool ----------------
__global__ __launch_bounds__(256) void k_pool(
    const float* __restrict__ xl, const float* __restrict__ g, const float* __restrict__ b,
    float* __restrict__ pooled)
{
    int i = blockIdx.x * 256 + threadIdx.x;
    const int total = 8 * CL * NS;
    if (i >= total) return;
    int s = i % NS;
    int c = (i / NS) % CL;
    int bb = i / (NS * CL);
    int gy = s / 12, gx = s % 12;
    const float* src = xl + ((long)bb * CL + c) * PXL + (gy * 8) * 96 + gx * 8;
    float sc = g[c] * rsqrtf(1.0f + EPS_BN);
    float sh = b[c];
    float m = -3.4e38f;
#pragma unroll
    for (int yy = 0; yy < 8; ++yy)
#pragma unroll
        for (int xx = 0; xx < 8; ++xx)
            m = fmaxf(m, src[yy * 96 + xx] * sc + sh);
    pooled[i] = m;
}

// ---------------- Kernel 4b: kv = kv_w * pooled + kv_b (tiny) ----------------
__global__ __launch_bounds__(256) void k_kv(
    const float* __restrict__ pooled, const float* __restrict__ w,
    const float* __restrict__ bias, float* __restrict__ kv)
{
    int i = blockIdx.x * 256 + threadIdx.x;
    const int total = 8 * 2 * CH * NS;
    if (i >= total) return;
    int s = i % NS;
    int o = (i / NS) % (2 * CH);
    int b = i / (NS * 2 * CH);
    const float* pb = pooled + (long)b * CL * NS;
    float acc = bias[o];
    for (int c = 0; c < CL; ++c) acc += w[o * CL + c] * pb[c * NS + s];
    kv[i] = acc;                              // [b][o(0..127)][s]
}

// ---------------- Kernel 5: fused attention + residual ----------------
__global__ __launch_bounds__(256) void k_attn(
    const float* __restrict__ xh1, const float* __restrict__ kv,
    const float* __restrict__ ng, const float* __restrict__ nb, float* __restrict__ out)
{
    __shared__ __bf16 ks[NS * CH];           // 18 KB  k_mat [s][c]
    __shared__ __bf16 vs[CH * NSP];          // 20 KB  v_mat [o][s], zero-padded K
    __shared__ float  Ls[8][NS * 16];        // 72 KB  logits per wave [s][pix]
    __shared__ __bf16 Ps[8][NSP * 16];       // 40 KB  probabilities per wave [s][pix]
    __shared__ float  sArr[CH], tArr[CH];

    const int blocksPerB = PXH / 128;        // 288
    int b = blockIdx.x / blocksPerB;
    int pbase = (blockIdx.x % blocksPerB) * 128;
    const float* kvb = kv + (long)b * 2 * CH * NS;

    for (int i = threadIdx.x; i < NS * CH; i += 256) {
        int s = i >> 6, c = i & 63;
        ks[i] = f2bf(kvb[c * NS + s]);
    }
    for (int i = threadIdx.x; i < CH * NSP; i += 256) {
        int o = i / NSP, s = i % NSP;
        vs[i] = f2bf(s < NS ? kvb[(CH + o) * NS + s] : 0.0f);
    }
    if (threadIdx.x < CH) {
        sArr[threadIdx.x] = ng[threadIdx.x] * rsqrtf(1.0f + EPS_BN);
        tArr[threadIdx.x] = nb[threadIdx.x];
    }
    __syncthreads();

    int wave = threadIdx.x >> 5, lane = threadIdx.x & 31;
    int n = lane & 15, hi = lane >> 4;
    int p = pbase + wave * 16;
    const float* xb = xh1 + (long)b * CH * PXH + p;
    float*       yb = out + (long)b * CH * PXH + p;
    float* Lw = Ls[wave];
    __bf16* Pw = Ps[wave];

    // hoisted xn_h fragments (each x_h element read+normalized once)
    v16bf bx[2];
#pragma unroll
    for (int kt = 0; kt < 2; ++kt) bx[kt] = b_frag_gmem_bn(xb, PXH, kt * 32, lane, sArr, tArr);

    // logits = K * bn(x_h) * S^-0.5
    for (int st = 0; st < 9; ++st) {
        v8f acc = {0.f, 0.f, 0.f, 0.f, 0.f, 0.f, 0.f, 0.f};
#pragma unroll
        for (int kt = 0; kt < 2; ++kt)
            acc = wmma_bf16(a_frag_lds(ks, st * 16, CH, kt * 32, lane), bx[kt], acc);
#pragma unroll
        for (int r = 0; r < 8; ++r)
            Lw[(st * 16 + r + 8 * hi) * 16 + n] = acc[r] * (1.0f / 12.0f);
    }
    __syncthreads();

    // per-pixel softmax (both halves compute stats for their pixel; half 0 writes P)
    float mx = -3.4e38f;
    for (int s = 0; s < NS; ++s) mx = fmaxf(mx, Lw[s * 16 + n]);
    float sum = 0.0f;
    for (int s = 0; s < NS; ++s) sum += __expf(Lw[s * 16 + n] - mx);
    float inv = 1.0f / sum;
    if (hi == 0) {
        for (int s = 0; s < NS; ++s) Pw[s * 16 + n] = f2bf(__expf(Lw[s * 16 + n] - mx) * inv);
        for (int s = NS; s < NSP; ++s) Pw[s * 16 + n] = f2bf(0.0f);
    }
    __syncthreads();

    // out = V * P + x_h
    v16bf bp[5];
#pragma unroll
    for (int kt = 0; kt < 5; ++kt) bp[kt] = b_frag_lds16(Pw, kt * 32, lane);

    for (int ot = 0; ot < 4; ++ot) {
        v8f acc = {0.f, 0.f, 0.f, 0.f, 0.f, 0.f, 0.f, 0.f};
#pragma unroll
        for (int kt = 0; kt < 5; ++kt)
            acc = wmma_bf16(a_frag_lds(vs, ot * 16, NSP, kt * 32, lane), bp[kt], acc);
#pragma unroll
        for (int r = 0; r < 8; ++r) {
            int m = ot * 16 + r + 8 * hi;
            long idx = (long)m * PXH + n;
            yb[idx] = xb[idx] + acc[r];
        }
    }
}

// ---------------- Kernel 6: x_h + MLP(x_h), Ch=64, in-place ----------------
__global__ __launch_bounds__(256) void k_mlp_h(
    const float* x, const float* w1, const float* b1,
    const float* w2, const float* b2, float* y)
{
    __shared__ __bf16 w1s[CH * CH];          // 8 KB
    __shared__ __bf16 w2s[CH * CH];          // 8 KB
    __shared__ __bf16 hs[8][CH * 16];        // 16 KB

    int tid = threadIdx.x;
    for (int i = tid; i < CH * CH; i += 256) { w1s[i] = f2bf(w1[i]); w2s[i] = f2bf(w2[i]); }
    __syncthreads();

    int wave = tid >> 5, lane = tid & 31;
    int n = lane & 15, hi = lane >> 4;
    long g0 = ((long)blockIdx.x * 8 + wave) * 16;
    int b = (int)(g0 / PXH);
    int p = (int)(g0 % PXH);
    const float* xb = x + (long)b * CH * PXH + p;
    float*       yb = y + (long)b * CH * PXH + p;
    __bf16* h = hs[wave];

    v16bf bx[2];
#pragma unroll
    for (int kt = 0; kt < 2; ++kt) bx[kt] = b_frag_gmem(xb, PXH, kt * 32, lane);

    for (int ot = 0; ot < 4; ++ot) {
        v8f acc;
#pragma unroll
        for (int r = 0; r < 8; ++r) acc[r] = b1[ot * 16 + r + 8 * hi];
#pragma unroll
        for (int kt = 0; kt < 2; ++kt)
            acc = wmma_bf16(a_frag_lds(w1s, ot * 16, CH, kt * 32, lane), bx[kt], acc);
#pragma unroll
        for (int r = 0; r < 8; ++r) {
            int m = ot * 16 + r + 8 * hi;
            h[m * 16 + n] = f2bf(gelu_exact(acc[r]));
        }
    }
    __syncthreads();

    v16bf bh[2];
#pragma unroll
    for (int kt = 0; kt < 2; ++kt) bh[kt] = b_frag_lds16(h, kt * 32, lane);

    for (int ot = 0; ot < 4; ++ot) {
        v8f acc;
#pragma unroll
        for (int r = 0; r < 8; ++r) acc[r] = b2[ot * 16 + r + 8 * hi];
#pragma unroll
        for (int kt = 0; kt < 2; ++kt)
            acc = wmma_bf16(a_frag_lds(w2s, ot * 16, CH, kt * 32, lane), bh[kt], acc);
#pragma unroll
        for (int r = 0; r < 8; ++r) {
            int m = ot * 16 + r + 8 * hi;
            long idx = (long)m * PXH + n;
            yb[idx] = acc[r] + xb[idx];
        }
    }
}

extern "C" void kernel_launch(void* const* d_in, const int* in_sizes, int n_in,
                              void* d_out, int out_size, void* d_ws, size_t ws_size,
                              hipStream_t stream) {
    (void)in_sizes; (void)n_in; (void)out_size; (void)ws_size;
    const float* x_l       = (const float*)d_in[0];
    const float* x_h       = (const float*)d_in[1];
    const float* mlp_l_w1  = (const float*)d_in[2];
    const float* mlp_l_b1  = (const float*)d_in[3];
    const float* mlp_l_w2  = (const float*)d_in[4];
    const float* mlp_l_b2  = (const float*)d_in[5];
    const float* conv_w    = (const float*)d_in[6];
    const float* conv_bn_g = (const float*)d_in[7];
    const float* conv_bn_b = (const float*)d_in[8];
    const float* kv_bn_g   = (const float*)d_in[9];
    const float* kv_bn_b   = (const float*)d_in[10];
    const float* kv_w      = (const float*)d_in[11];
    const float* kv_b      = (const float*)d_in[12];
    const float* norm_g    = (const float*)d_in[13];
    const float* norm_b    = (const float*)d_in[14];
    const float* mlp_h_w1  = (const float*)d_in[15];
    const float* mlp_h_b1  = (const float*)d_in[16];
    const float* mlp_h_w2  = (const float*)d_in[17];
    const float* mlp_h_b2  = (const float*)d_in[18];
    float* out = (float*)d_out;
    float* ws  = (float*)d_ws;

    float* xl_new = ws;                                  //  9,437,184 f
    float* xlc    = ws + 9437184;                        //  4,718,592 f
    float* xh1    = ws + 14155776;                       // 18,874,368 f
    float* pooled = ws + 33030144;                       //    147,456 f
    float* kvbuf  = ws + 33177600;                       //    147,456 f

    k_mlp_l       <<<576,  256, 0, stream>>>(x_l, mlp_l_w1, mlp_l_b1, mlp_l_w2, mlp_l_b2, xl_new);
    k_conv_bn_relu<<<576,  256, 0, stream>>>(xl_new, conv_w, conv_bn_g, conv_bn_b, xlc);
    k_up_add      <<<73728, 256, 0, stream>>>(x_h, xlc, xh1);
    k_pool        <<<(8 * CL * NS + 255) / 256, 256, 0, stream>>>(xl_new, kv_bn_g, kv_bn_b, pooled);
    k_kv          <<<(8 * 2 * CH * NS + 255) / 256, 256, 0, stream>>>(pooled, kv_w, kv_b, kvbuf);
    k_attn        <<<2304, 256, 0, stream>>>(xh1, kvbuf, norm_g, norm_b, out);
    k_mlp_h       <<<2304, 256, 0, stream>>>(out, mlp_h_w1, mlp_h_b1, mlp_h_w2, mlp_h_b2, out);
}
